// DGCNNSegmentation_22479858828029
// MI455X (gfx1250) — matrix operand order repaired and checked by reading, not compiled, not run on hardware
//
#include <hip/hip_runtime.h>

// ---------------------------------------------------------------------------
// DGCNN segmentation for MI455X (gfx1250, wave32, WMMA).
//
// EdgeConv factorization (g>0, LReLU monotone => max commutes with the
// affine+activation):
//     s = W_left @ h,  t = (W_right - W_left) @ h
//     y[o,n] = LReLU(g'[o] * (max_k s[o, idx[n,k]] + t[o,n]) + b[o])
//
// Dominant compute = pairwise-distance Gram (~36 GFLOP).  Done with
// error-compensated split-bf16 on V_WMMA_F32_16X16X32_BF16:
//     h = hi + lo  (bf16 each),   h.h' ~= hi.hi' + hi.lo' + lo.hi'
// -> 3 K=32 WMMAs replace 8 K=4 fp32 WMMAs with ~fp32 product accuracy.
// h is pre-packed into WMMA-native A(16x32)/B(32x16) fragment layouts so each
// lane loads a full v16bf fragment as one 32B (2 x b128) load.
// Small GEMMs (Wst / head MLP) stay on exact V_WMMA_F32_16X16X4_F32.
// Distance chunks (32 MB) are L2-resident (192 MB L2).
// ---------------------------------------------------------------------------

#define NPTS   4096
#define BATCH  4
#define KNN    20
#define RCHUNK 2048
#define C0PAD  12              // layer-1 input channels padded 9 -> 12
#define NBLKMAX 4              // max K blocks of 32 channels (layer 4: 128)
#define INV_SQRT_1PEPS 0.99999500003749975f
#define LRELU_NEG 0.2f

typedef __attribute__((ext_vector_type(2)))  float  v2f;
typedef __attribute__((ext_vector_type(8)))  float  v8f;
typedef __attribute__((ext_vector_type(16))) __bf16 v16bf;

__device__ __forceinline__ float lrelu(float v) { return v >= 0.f ? v : LRELU_NEG * v; }

#define WMMA_F32(A_, B_, C_) \
  __builtin_amdgcn_wmma_f32_16x16x4_f32(false, (A_), false, (B_), (short)0, (C_), false, false)
#define WMMA_BF16(A_, B_, C_) \
  __builtin_amdgcn_wmma_f32_16x16x32_bf16(false, (A_), false, (B_), (short)0, (C_), false, false)

// ---------------- input transpose + zero-pad: x (B,N,9) -> h0 (B,12,N) ------
__global__ __launch_bounds__(256) void k_transpose_in(const float* __restrict__ x,
                                                      float* __restrict__ h0) {
  int i = blockIdx.x * blockDim.x + threadIdx.x;
  const int total = BATCH * NPTS * C0PAD;
  if (i >= total) return;
  int c = i % C0PAD;
  int n = (i / C0PAD) % NPTS;
  int b = i / (C0PAD * NPTS);
  float v = (c < 9) ? x[((size_t)b * NPTS + n) * 9 + c] : 0.f;
  h0[((size_t)b * C0PAD + c) * NPTS + n] = v;
}

// ---------------- per-point squared norm (exact fp32) -----------------------
__global__ __launch_bounds__(256) void k_rowsq(const float* __restrict__ h,
                                               float* __restrict__ sq, int C) {
  int n = blockIdx.x * blockDim.x + threadIdx.x;
  if (n >= NPTS) return;
  float s = 0.f;
  for (int c = 0; c < C; ++c) { float v = h[(size_t)c * NPTS + n]; s += v * v; }
  sq[n] = s;
}

// ---------------- pack h into split-bf16 WMMA fragment layouts --------------
// For K-block blk (32 channels) and lane-half `half`:
//   A-frag lane(l15,half) needs channels base + half*8 + {0..7, 16..23}
//   B-frag lane(l15,half) needs channels base + half*8 + {0..7, 16..23}
// One thread per (blk, half, n): writes 16 bf16 (hi and lo) per layout.
__global__ __launch_bounds__(256) void k_pack_bf16(const float* __restrict__ h,
                                                   int rows, int nblk,
                                                   __bf16* __restrict__ pAh,
                                                   __bf16* __restrict__ pAl,
                                                   __bf16* __restrict__ pBh,
                                                   __bf16* __restrict__ pBl) {
  int i = blockIdx.x * blockDim.x + threadIdx.x;
  int total = nblk * 2 * NPTS;
  if (i >= total) return;
  int n    = i % NPTS;
  int half = (i / NPTS) & 1;
  int blk  = i / (2 * NPTS);
  int base = blk * 32;
  v16bf ah, al, bh, bl;
#pragma unroll
  for (int t = 0; t < 16; ++t) {
    int ca = base + half * 8 + ((t < 8) ? t : (8 + t));   // {0..7} then {16..23}
    int cb = base + half * 8 + ((t < 8) ? t : (8 + t));
    float va = (ca < rows) ? h[(size_t)ca * NPTS + n] : 0.f;
    float vb = (cb < rows) ? h[(size_t)cb * NPTS + n] : 0.f;
    __bf16 vah = (__bf16)va;
    __bf16 vbh = (__bf16)vb;
    ah[t] = vah;  al[t] = (__bf16)(va - (float)vah);
    bh[t] = vbh;  bl[t] = (__bf16)(vb - (float)vbh);
  }
  size_t off = ((size_t)(blk * 2 + half) * NPTS + n) * 16;
  *(v16bf*)(pAh + off) = ah;
  *(v16bf*)(pAl + off) = al;
  *(v16bf*)(pBh + off) = bh;
  *(v16bf*)(pBl + off) = bl;
}

// ---------------- Gram/distance: split-bf16 WMMA, 16x64 strip per wave ------
// dist[i,j] = 2*h_i.h_j - |h_i|^2 - |h_j|^2 with compensated bf16 products.
__global__ __launch_bounds__(256) void k_gram_wmma(const __bf16* __restrict__ pAh,
                                                   const __bf16* __restrict__ pAl,
                                                   const __bf16* __restrict__ pBh,
                                                   const __bf16* __restrict__ pBl,
                                                   const float* __restrict__ sq,
                                                   float* __restrict__ dist,
                                                   int nblk, int r0) {
  int wave = blockIdx.x * (blockDim.x >> 5) + (threadIdx.x >> 5);
  int lane = threadIdx.x & 31;
  const int stripsJ = NPTS / 64;
  int ti = wave / stripsJ, tj = wave % stripsJ;
  int i0 = r0 + ti * 16, j0 = tj * 64;
  int l15 = lane & 15, hi = lane >> 4;

  size_t aoff = ((size_t)hi * NPTS + i0 + l15) * 16;
  size_t boff = ((size_t)hi * NPTS + j0 + l15) * 16;
  const size_t blkstep = (size_t)2 * NPTS * 16;

  v8f acc0 = {}, acc1 = {}, acc2 = {}, acc3 = {};
  for (int blk = 0; blk < nblk; ++blk) {
    v16bf aH  = *(const v16bf*)(pAh + aoff);
    v16bf aL  = *(const v16bf*)(pAl + aoff);
    v16bf b0H = *(const v16bf*)(pBh + boff);
    v16bf b1H = *(const v16bf*)(pBh + boff + 16 * 16);
    v16bf b2H = *(const v16bf*)(pBh + boff + 32 * 16);
    v16bf b3H = *(const v16bf*)(pBh + boff + 48 * 16);
    v16bf b0L = *(const v16bf*)(pBl + boff);
    v16bf b1L = *(const v16bf*)(pBl + boff + 16 * 16);
    v16bf b2L = *(const v16bf*)(pBl + boff + 32 * 16);
    v16bf b3L = *(const v16bf*)(pBl + boff + 48 * 16);
    // interleave so dependent accumulates are >= 4 WMMAs apart
    acc0 = WMMA_BF16(aH, b0H, acc0);
    acc1 = WMMA_BF16(aH, b1H, acc1);
    acc2 = WMMA_BF16(aH, b2H, acc2);
    acc3 = WMMA_BF16(aH, b3H, acc3);
    acc0 = WMMA_BF16(aH, b0L, acc0);
    acc1 = WMMA_BF16(aH, b1L, acc1);
    acc2 = WMMA_BF16(aH, b2L, acc2);
    acc3 = WMMA_BF16(aH, b3L, acc3);
    acc0 = WMMA_BF16(aL, b0H, acc0);
    acc1 = WMMA_BF16(aL, b1H, acc1);
    acc2 = WMMA_BF16(aL, b2H, acc2);
    acc3 = WMMA_BF16(aL, b3H, acc3);
    aoff += blkstep;  boff += blkstep;
  }

  float sq0 = sq[j0 + l15];
  float sq1 = sq[j0 + 16 + l15];
  float sq2 = sq[j0 + 32 + l15];
  float sq3 = sq[j0 + 48 + l15];
#pragma unroll
  for (int v = 0; v < 8; ++v) {
    int irow = i0 + v + (hi << 3);
    float si = sq[irow];
    float* dr = dist + (size_t)(irow - r0) * NPTS + l15;
    dr[j0]      = 2.f * acc0[v] - si - sq0;
    dr[j0 + 16] = 2.f * acc1[v] - si - sq1;
    dr[j0 + 32] = 2.f * acc2[v] - si - sq2;
    dr[j0 + 48] = 2.f * acc3[v] - si - sq3;
  }
}

// ---------------- top-K per row: one wave32 per row -------------------------
__global__ __launch_bounds__(256) void k_topk(const float* __restrict__ dist,
                                              int* __restrict__ idxOut, int r0) {
  int lane = threadIdx.x & 31;
  int row  = blockIdx.x * (blockDim.x >> 5) + (threadIdx.x >> 5);
  const float* drow = dist + (size_t)row * NPTS;

  float vals[KNN]; int ids[KNN];
#pragma unroll
  for (int t = 0; t < KNN; ++t) { vals[t] = -3.4e38f; ids[t] = 0; }

  for (int j = lane * 4; j < NPTS; j += 128) {
    float4 q = *(const float4*)(drow + j);
    float e[4] = {q.x, q.y, q.z, q.w};
#pragma unroll
    for (int u = 0; u < 4; ++u) {
      float v = e[u];
      if (v > vals[KNN - 1]) {
        int id = j + u;
#pragma unroll
        for (int t = 0; t < KNN; ++t) {
          bool gt = v > vals[t];
          float tv = vals[t]; int ti2 = ids[t];
          vals[t] = gt ? v : tv;  ids[t] = gt ? id : ti2;
          v = gt ? tv : v;        id = gt ? ti2 : id;
        }
      }
    }
  }

  int keep = 0;
#pragma unroll
  for (int r = 0; r < KNN; ++r) {
    float v = vals[0]; int id = ids[0]; int ln = lane;
#pragma unroll
    for (int off = 16; off >= 1; off >>= 1) {
      float ov = __shfl_xor(v, off, 32);
      int  oid = __shfl_xor(id, off, 32);
      int  oln = __shfl_xor(ln, off, 32);
      if (ov > v || (ov == v && oln < ln)) { v = ov; id = oid; ln = oln; }
    }
    bool win = (ln == lane);
#pragma unroll
    for (int t = 0; t < KNN - 1; ++t) {
      vals[t] = win ? vals[t + 1] : vals[t];
      ids[t]  = win ? ids[t + 1]  : ids[t];
    }
    if (win) vals[KNN - 1] = -3.4e38f;
    if (lane == r) keep = id;
  }
  if (lane < KNN) idxOut[(size_t)(r0 + row) * KNN + lane] = keep;
}

// ---------------- fp32 WMMA GEMM, 16x64 strip per wave, fused epilogue ------
// C[M x NPTS] = A[M x K row-major] @ B[K x NPTS].  M%16==0, K%4==0: no guards.
__global__ __launch_bounds__(256) void k_gemm_wmma(const float* __restrict__ A,
                                                   const float* __restrict__ Bm,
                                                   float* __restrict__ Cout,
                                                   int Kc,
                                                   const float* __restrict__ gamma,
                                                   const float* __restrict__ beta,
                                                   int act) {
  int wave = blockIdx.x * (blockDim.x >> 5) + (threadIdx.x >> 5);
  int lane = threadIdx.x & 31;
  const int stripsJ = NPTS / 64;
  int tm = wave / stripsJ, tj = wave % stripsJ;
  int m0 = tm * 16, j0 = tj * 64;
  int l15 = lane & 15, hi = lane >> 4;

  const float* pa = A + (size_t)(m0 + l15) * Kc + (hi << 1);
  const float* pb = Bm + (size_t)hi * NPTS + j0 + l15;

  v8f acc0 = {}, acc1 = {}, acc2 = {}, acc3 = {};
  v2f a, b0, b1, b2, b3;
  a.x  = pa[0];        a.y  = pa[1];
  b0.x = pb[0];        b0.y = pb[2 * NPTS];
  b1.x = pb[16];       b1.y = pb[2 * NPTS + 16];
  b2.x = pb[32];       b2.y = pb[2 * NPTS + 32];
  b3.x = pb[48];       b3.y = pb[2 * NPTS + 48];

  for (int kb = 4; kb < Kc; kb += 4) {
    pa += 4;  pb += (size_t)4 * NPTS;
    v2f na, nb0, nb1, nb2, nb3;
    na.x  = pa[0];      na.y  = pa[1];
    nb0.x = pb[0];      nb0.y = pb[2 * NPTS];
    nb1.x = pb[16];     nb1.y = pb[2 * NPTS + 16];
    nb2.x = pb[32];     nb2.y = pb[2 * NPTS + 32];
    nb3.x = pb[48];     nb3.y = pb[2 * NPTS + 48];
    acc0 = WMMA_F32(a, b0, acc0);
    acc1 = WMMA_F32(a, b1, acc1);
    acc2 = WMMA_F32(a, b2, acc2);
    acc3 = WMMA_F32(a, b3, acc3);
    a = na; b0 = nb0; b1 = nb1; b2 = nb2; b3 = nb3;
  }
  acc0 = WMMA_F32(a, b0, acc0);
  acc1 = WMMA_F32(a, b1, acc1);
  acc2 = WMMA_F32(a, b2, acc2);
  acc3 = WMMA_F32(a, b3, acc3);

#pragma unroll
  for (int v = 0; v < 8; ++v) {
    int m = m0 + v + (hi << 3);
    float gs = gamma ? (gamma[m] * INV_SQRT_1PEPS) : 1.f;
    float bs = beta ? beta[m] : 0.f;
    float* cr = Cout + (size_t)m * NPTS + j0 + l15;
    float r0v = acc0[v] * gs + bs;
    float r1v = acc1[v] * gs + bs;
    float r2v = acc2[v] * gs + bs;
    float r3v = acc3[v] * gs + bs;
    if (act) { r0v = lrelu(r0v); r1v = lrelu(r1v); r2v = lrelu(r2v); r3v = lrelu(r3v); }
    cr[0] = r0v; cr[16] = r1v; cr[32] = r2v; cr[48] = r3v;
  }
}

// ---------------- build stacked [W_left ; W_right - W_left], K padded -------
__global__ __launch_bounds__(256) void k_build_wst(const float* __restrict__ W,
                                                   float* __restrict__ Wst,
                                                   int outC, int Cin, int Cpad) {
  int i = blockIdx.x * blockDim.x + threadIdx.x;
  int total = outC * Cpad;
  if (i >= total) return;
  int o = i / Cpad, c = i % Cpad;
  float wl = (c < Cin) ? W[(size_t)o * (2 * Cin) + c]       : 0.f;
  float wr = (c < Cin) ? W[(size_t)o * (2 * Cin) + Cin + c] : 0.f;
  Wst[(size_t)o * Cpad + c] = wl;
  Wst[((size_t)outC + o) * Cpad + c] = wr - wl;
}

// ---------------- pad head layer-3 weights 13x128 -> 16x128, bias -> 16 -----
__global__ __launch_bounds__(256) void k_pad_head(const float* __restrict__ Wc3,
                                                  const float* __restrict__ bc3,
                                                  float* __restrict__ Wp,
                                                  float* __restrict__ bp) {
  int i = blockIdx.x * blockDim.x + threadIdx.x;
  if (i < 16 * 128) {
    int o = i / 128, c = i % 128;
    Wp[i] = (o < 13) ? Wc3[(size_t)o * 128 + c] : 0.f;
  }
  if (i < 16) bp[i] = (i < 13) ? bc3[i] : 0.f;
}

// ---------------- gather-max + affine + LeakyReLU ---------------------------
__global__ __launch_bounds__(256) void k_edge_max(const float* __restrict__ st,
                                                  long long stStride,
                                                  const int* __restrict__ idx,
                                                  const float* __restrict__ g,
                                                  const float* __restrict__ bias,
                                                  float* __restrict__ out,
                                                  long long outStride, int outC) {
  int n = blockIdx.x * blockDim.x + threadIdx.x;
  int o = blockIdx.y;
  int b = blockIdx.z;
  const float* s = st + (long long)b * stStride + (size_t)o * NPTS;
  const float* t = st + (long long)b * stStride + (size_t)(outC + o) * NPTS;
  const int4* ib4 = (const int4*)(idx + ((size_t)b * NPTS + n) * KNN);
  float m = -3.4e38f;
#pragma unroll
  for (int q = 0; q < KNN / 4; ++q) {
    int4 v4 = ib4[q];
    m = fmaxf(m, s[v4.x]); m = fmaxf(m, s[v4.y]);
    m = fmaxf(m, s[v4.z]); m = fmaxf(m, s[v4.w]);
  }
  float val = (m + t[n]) * (g[o] * INV_SQRT_1PEPS) + bias[o];
  out[(long long)b * outStride + (size_t)o * NPTS + n] = lrelu(val);
}

// ---------------- final output transpose (B,16,N) -> (B,N,13) ---------------
__global__ __launch_bounds__(256) void k_out_transpose(const float* __restrict__ z3,
                                                       float* __restrict__ out) {
  int i = blockIdx.x * blockDim.x + threadIdx.x;
  const int total = BATCH * NPTS * 13;
  if (i >= total) return;
  int o = i % 13;
  int n = (i / 13) % NPTS;
  int b = i / (13 * NPTS);
  out[i] = z3[((size_t)b * 16 + o) * NPTS + n];
}

// ===========================================================================
extern "C" void kernel_launch(void* const* d_in, const int* in_sizes, int n_in,
                              void* d_out, int out_size, void* d_ws, size_t ws_size,
                              hipStream_t stream) {
  (void)in_sizes; (void)n_in; (void)out_size; (void)ws_size;
  const float* x    = (const float*)d_in[0];
  const float* W[4]  = {(const float*)d_in[1], (const float*)d_in[4],
                        (const float*)d_in[7], (const float*)d_in[10]};
  const float* g[4]  = {(const float*)d_in[2], (const float*)d_in[5],
                        (const float*)d_in[8], (const float*)d_in[11]};
  const float* bb[4] = {(const float*)d_in[3], (const float*)d_in[6],
                        (const float*)d_in[9], (const float*)d_in[12]};
  const float* Wc1 = (const float*)d_in[13];
  const float* gc1 = (const float*)d_in[14];
  const float* bc1 = (const float*)d_in[15];
  const float* Wc2 = (const float*)d_in[16];
  const float* gc2 = (const float*)d_in[17];
  const float* bc2 = (const float*)d_in[18];
  const float* Wc3 = (const float*)d_in[19];
  const float* bc3 = (const float*)d_in[20];

  // ---- workspace carve (256B aligned) ----
  char* p = (char*)d_ws;
  auto allocB = [&](size_t bytes) -> void* {
    void* r = (void*)p;
    p += ((bytes + 255) / 256) * 256;
    return r;
  };
  float* h0    = (float*)allocB((size_t)BATCH * C0PAD * NPTS * 4);
  float* xcat  = (float*)allocB((size_t)BATCH * 512 * NPTS * 4);
  float* sq    = (float*)allocB((size_t)NPTS * 4);
  int*   idx   = (int*)allocB((size_t)BATCH * NPTS * KNN * 4);
  float* Wst   = (float*)allocB((size_t)2 * 256 * 128 * 4);
  float* st    = (float*)allocB((size_t)BATCH * 2 * 256 * NPTS * 4);
  float* dist  = (float*)allocB((size_t)RCHUNK * NPTS * 4);
  float* z1    = (float*)allocB((size_t)BATCH * 256 * NPTS * 4);
  float* z2    = (float*)allocB((size_t)BATCH * 128 * NPTS * 4);
  float* z3    = (float*)allocB((size_t)BATCH * 16 * NPTS * 4);
  float* Wc3p  = (float*)allocB((size_t)16 * 128 * 4);
  float* bc3p  = (float*)allocB(16 * 4);
  const size_t packElems = (size_t)NBLKMAX * 2 * NPTS * 16;   // bf16 elements
  __bf16* pAh = (__bf16*)allocB(packElems * 2);
  __bf16* pAl = (__bf16*)allocB(packElems * 2);
  __bf16* pBh = (__bf16*)allocB(packElems * 2);
  __bf16* pBl = (__bf16*)allocB(packElems * 2);

  k_transpose_in<<<(BATCH * NPTS * C0PAD + 255) / 256, 256, 0, stream>>>(x, h0);
  k_pad_head<<<8, 256, 0, stream>>>(Wc3, bc3, Wc3p, bc3p);

  struct Lcfg { int Cin, Cpad, nblk, Cout, inOff, outOff; };
  const Lcfg L[4] = { {9, C0PAD, 1, 64, -1, 0}, {64, 64, 2, 64, 0, 64},
                      {64, 64, 2, 128, 64, 128}, {128, 128, 4, 256, 128, 256} };

  const long long stStride  = (long long)2 * 256 * NPTS;
  const long long catStride = (long long)512 * NPTS;

  for (int l = 0; l < 4; ++l) {
    int Cpad = L[l].Cpad, Cout = L[l].Cout, nblk = L[l].nblk;
    k_build_wst<<<(Cout * Cpad + 255) / 256, 256, 0, stream>>>(
        W[l], Wst, Cout, L[l].Cin, Cpad);

    for (int b = 0; b < BATCH; ++b) {
      const float* hb = (l == 0)
          ? h0 + (size_t)b * C0PAD * NPTS
          : xcat + ((size_t)b * 512 + L[l].inOff) * NPTS;
      int rows = (l == 0) ? C0PAD : L[l].Cin;

      k_rowsq<<<NPTS / 256, 256, 0, stream>>>(hb, sq, rows);
      k_pack_bf16<<<(nblk * 2 * NPTS + 255) / 256, 256, 0, stream>>>(
          hb, rows, nblk, pAh, pAl, pBh, pBl);

      for (int r0 = 0; r0 < NPTS; r0 += RCHUNK) {
        int waves = (RCHUNK / 16) * (NPTS / 64);       // 8192 waves
        k_gram_wmma<<<waves / 8, 256, 0, stream>>>(pAh, pAl, pBh, pBl,
                                                   sq, dist, nblk, r0);
        k_topk<<<RCHUNK / 8, 256, 0, stream>>>(dist, idx + (size_t)b * NPTS * KNN, r0);
      }

      int M = 2 * Cout;                                 // multiple of 16
      int waves = (M / 16) * (NPTS / 64);
      k_gemm_wmma<<<waves / 8, 256, 0, stream>>>(
          Wst, hb, st + (size_t)b * 2 * 256 * NPTS, Cpad, nullptr, nullptr, 0);
    }

    k_edge_max<<<dim3(NPTS / 256, Cout, BATCH), 256, 0, stream>>>(
        st, stStride, idx, g[l], bb[l],
        xcat + (size_t)L[l].outOff * NPTS, catStride, Cout);
  }

  // ---- head MLP: 512 -> 256 -> 128 -> 13(pad 16) ----
  for (int b = 0; b < BATCH; ++b) {
    const float* xb = xcat + (size_t)b * 512 * NPTS;
    float* z1b = z1 + (size_t)b * 256 * NPTS;
    float* z2b = z2 + (size_t)b * 128 * NPTS;
    float* z3b = z3 + (size_t)b * 16 * NPTS;
    k_gemm_wmma<<<(256 / 16) * (NPTS / 64) / 8, 256, 0, stream>>>(
        Wc1, xb, z1b, 512, gc1, bc1, 1);
    k_gemm_wmma<<<(128 / 16) * (NPTS / 64) / 8, 256, 0, stream>>>(
        Wc2, z1b, z2b, 256, gc2, bc2, 1);
    k_gemm_wmma<<<(16 / 16) * (NPTS / 64) / 8, 256, 0, stream>>>(
        Wc3p, z2b, z3b, 128, nullptr, bc3p, 0);
  }

  k_out_transpose<<<(BATCH * NPTS * 13 + 255) / 256, 256, 0, stream>>>(z3, (float*)d_out);
}
